// SimpleVanillaRFRNN_62543313764708
// MI455X (gfx1250) — matrix operand order repaired and checked by reading, not compiled, not run on hardware
//
#include <hip/hip_runtime.h>
#include <math.h>

// ---------------- problem constants ----------------
constexpr int T = 256, B = 128, I = 256, H = 1024, O = 32;
constexpr float DT = 0.01f, THETA = 1.0f;

// output layout (flat f32): outs[T,B,O], zf[B,H], uf[B,H], ouf[B,O], ssum
constexpr size_t OUTS_OFF = 0;
constexpr size_t ZF_OFF   = (size_t)T * B * O;              // 1048576
constexpr size_t UF_OFF   = ZF_OFF + (size_t)B * H;         // 1179648
constexpr size_t OUF_OFF  = UF_OFF + (size_t)B * H;         // 1310720
constexpr size_t SSUM_OFF = OUF_OFF + (size_t)B * O;        // 1314816

// ---------------- workspace layout (bytes) ----------------
constexpr size_t ZELEMS   = (size_t)B * H;             // one z buffer (f16 elems)
constexpr size_t SYNC_OFF = 0;                         // arrive,release,spikecnt (+pad)
constexpr size_t ZBUF_OFF = 256;                       // f16 z, PING-PONG: [2][B,H]
constexpr size_t ZBUF_BYTES = 2 * ZELEMS * 2;          // 512 KB
constexpr size_t WXP_OFF  = ZBUF_OFF + ZBUF_BYTES;     // packed Wx  f16 [8][64][32][16]
constexpr size_t WXP_BYTES = (size_t)I * H * 2;
constexpr size_t WZP_OFF  = WXP_OFF + WXP_BYTES;       // packed Wz  f16 [32][64][32][16]
constexpr size_t WZP_BYTES = (size_t)H * H * 2;
constexpr size_t WOP_OFF  = WZP_OFF + WZP_BYTES;       // packed Wo  f16 [32][2][32][16]
constexpr size_t WOP_BYTES = (size_t)H * O * 2;
// xcur stored in C-fragment layout: [T*8 mtiles][64 ntiles][32 lanes][8] f16
constexpr size_t XCUR_OFF = WOP_OFF + WOP_BYTES;
constexpr size_t XCUR_BYTES = (size_t)T * B * H * 2;   // 64 MB

typedef _Float16 v16h __attribute__((ext_vector_type(16)));
typedef _Float16 v8h  __attribute__((ext_vector_type(8)));
typedef float    v8f  __attribute__((ext_vector_type(8)));

__device__ __forceinline__ v8f wmma_f16(v16h a, v16h b, v8f c) {
  return __builtin_amdgcn_wmma_f32_16x16x32_f16(false, a, false, b, (short)0, c,
                                                false, false);
}

__device__ __forceinline__ v16h cat8(v8h lo, v8h hi) {
  return __builtin_shufflevector(lo, hi, 0, 1, 2, 3, 4, 5, 6, 7,
                                 8, 9, 10, 11, 12, 13, 14, 15);
}

// 16x32 f16 A-fragment from a row-major f16 row pointer (this lane's row):
// K comes as two contiguous 8-element runs (CDNA5 16-bit A layout).
__device__ __forceinline__ v16h load_a_frag(const _Float16* rowptr, int kbase, int s) {
  v8h lo = *(const v8h*)(rowptr + kbase + 8 * s);
  v8h hi = *(const v8h*)(rowptr + kbase + 16 + 8 * s);
  return cat8(lo, hi);
}

// CDNA5 async copy: 16 bytes/lane global -> LDS, tracked by ASYNCcnt.
__device__ __forceinline__ void async_copy_b128(unsigned lds_byte_off, const void* gaddr) {
  asm volatile("global_load_async_to_lds_b128 %0, %1, off"
               :: "v"(lds_byte_off), "v"(gaddr) : "memory");
}
__device__ __forceinline__ void wait_async_zero() {
  asm volatile("s_wait_asynccnt 0x0" ::: "memory");
}

// K/32-tile GEMM over packed-B weights with explicit 2-deep B double buffer.
// bload(kt) returns packed B fragment for K-tile kt; aload(kt) the A fragment.
template <int KT, typename ALoad, typename BLoad>
__device__ __forceinline__ v8f gemm_pipe(ALoad aload, BLoad bload) {
  v8f acc = {};
  v16h b0 = bload(0);
  v16h b1 = bload(1);
#pragma unroll 4
  for (int kt = 0; kt < KT; kt += 2) {
    acc = wmma_f16(aload(kt), b0, acc);
    if (kt + 2 < KT) b0 = bload(kt + 2);
    acc = wmma_f16(aload(kt + 1), b1, acc);
    if (kt + 3 < KT) b1 = bload(kt + 3);
  }
  return acc;
}

// ---------------- kernel 1: pack f32 weights into WMMA B-operand layout ------
__global__ void pack_w_kernel(const float* __restrict__ W, _Float16* __restrict__ dst,
                              int rowOff, int NT, int total, int ldw) {
  int idx = blockIdx.x * 256 + threadIdx.x;
  if (idx >= total) return;
  int e    = idx & 15;
  int lane = (idx >> 4) & 31;
  int nt   = (idx >> 9) % NT;
  int kt   = idx / (512 * NT);
  int k = kt * 32 + 8 * (lane >> 4) + (e & 7) + 16 * (e >> 3);
  int n = nt * 16 + (lane & 15);
  dst[idx] = (_Float16)W[(size_t)(rowOff + k) * ldw + n];
}

// ---------------- kernel 2: xcur = x_t @ W_x (parallel over T*B) -------------
__global__ void __launch_bounds__(256) xcur_gemm_kernel(
    const float* __restrict__ x, const _Float16* __restrict__ Wxp,
    _Float16* __restrict__ xcur) {
  const int lane = threadIdx.x & 31;
  const int wave = threadIdx.x >> 5;
  const int s = lane >> 4, col = lane & 15;
  const int mtile = blockIdx.x;               // over T*B/16
  const int arow = mtile * 16 + col;          // A row for this lane

  v16h afr[8];
#pragma unroll
  for (int kt = 0; kt < 8; ++kt) {
    const float* xa = x + (size_t)arow * I + kt * 32 + 8 * s;
    v16h a;
#pragma unroll
    for (int j = 0; j < 8; ++j) {
      a[j]     = (_Float16)xa[j];
      a[8 + j] = (_Float16)xa[16 + j];
    }
    afr[kt] = a;
  }

  for (int i = 0; i < 8; ++i) {
    const int nt = wave * 8 + i;
    const _Float16* bbase = Wxp + ((size_t)nt * 32 + lane) * 16;
    v8f acc = gemm_pipe<8>(
        [&](int kt) { return afr[kt]; },
        [&](int kt) { return *(const v16h*)(bbase + (size_t)kt * 64 * 512); });
    v8h h;
#pragma unroll
    for (int r = 0; r < 8; ++r) h[r] = (_Float16)acc[r];
    *(v8h*)(xcur + (((size_t)mtile * 64 + nt) * 32 + lane) * 8) = h;
  }
}

// ---------------- grid-wide barrier (monotonic, replay-safe) ----------------
__device__ __forceinline__ void grid_sync(unsigned* arrive, unsigned* release,
                                          unsigned nblocks, unsigned* phase) {
  __syncthreads();
  if (threadIdx.x == 0) {
    __threadfence();
    unsigned target = *phase + 1u;
    unsigned a = atomicAdd(arrive, 1u) + 1u;
    if (a == nblocks * target) {
      atomicExch(release, target);
    } else {
      while (atomicAdd(release, 0u) < target) __builtin_amdgcn_s_sleep(1);
    }
    __threadfence();
  }
  __syncthreads();
  *phase += 1u;
}

// ---------------- kernel 3: persistent recurrent core -----------------------
// Ping-pong z buffers: iter t reads zb[t&1], writes z_{t+1} to zb[(t+1)&1]
// -> ONE grid sync per timestep.
// blocks 0..63 ("cur"): mtile = blk>>3 (shared by the block's 8 waves; the 16
//   z rows are staged to LDS per step via async global->LDS), ntile =
//   (blk&7)*8 + wave; u,v state registers-resident for all T steps.
// blocks 64..65 ("readout"): otile = blk-64, mtile = wave; ou in registers.
__global__ void __launch_bounds__(256) rf_recurrent_kernel(
    const float* __restrict__ omega, const float* __restrict__ b_offset,
    const float* __restrict__ tau, const _Float16* __restrict__ Wzp,
    const _Float16* __restrict__ Wop, const _Float16* __restrict__ xcur,
    _Float16* __restrict__ zb, float* __restrict__ out,
    unsigned* __restrict__ sync_words, int nblocks) {
  __shared__ _Float16 zt[16 * H];   // 32 KB: this mtile's 16 z rows

  unsigned* arrive   = sync_words + 0;
  unsigned* release  = sync_words + 1;
  unsigned* spikecnt = sync_words + 2;

  const int lane = threadIdx.x & 31;
  const int wave = threadIdx.x >> 5;
  const int s = lane >> 4, col = lane & 15;
  unsigned phase = 0;

  const bool is_cur = (blockIdx.x < 64);
  int mtile, ntile;
  if (is_cur) { mtile = blockIdx.x >> 3; ntile = (blockIdx.x & 7) * 8 + wave; }
  else        { mtile = wave;            ntile = blockIdx.x - 64; }

  const int n    = ntile * 16 + col;  // h column (cur) or o column (readout)
  const int mrow = mtile * 16;        // base batch row
  const _Float16* zrow_l = zt + (size_t)col * H;          // LDS A row
  const unsigned  zt_base = (unsigned)(size_t)(&zt[0]);   // LDS byte offset
  const _Float16* wz_base = Wzp + ((size_t)ntile * 32 + lane) * 16;
  const _Float16* wo_base = Wop + ((size_t)ntile * 32 + lane) * 16;

  float om = 0.f, bb = 0.f, alpha = 0.f, oma = 0.f;
  if (is_cur) { om = omega[n]; bb = -b_offset[n]; }
  else        { alpha = expf(-DT / tau[n]); oma = 1.f - alpha; }

  v8f u = {}, v = {}, ou = {};
  unsigned cnt = 0;

  for (int t = 0; t < T; ++t) {
    const _Float16* zsrc = zb + (size_t)(t & 1) * ZELEMS;        // z_t
    _Float16*       zdst = zb + (size_t)((t + 1) & 1) * ZELEMS;  // z_{t+1}
    if (is_cur) {
      // stage this mtile's 16 z_t rows (32 KB) into LDS: 8 x 16B per thread
      const char* gsrc = (const char*)(zsrc + (size_t)mrow * H);
#pragma unroll
      for (int i = 0; i < 8; ++i) {
        unsigned off = (unsigned)(threadIdx.x + i * 256) * 16u;
        async_copy_b128(zt_base + off, gsrc + off);
      }
      wait_async_zero();
      __syncthreads();

      v8f acc = gemm_pipe<H / 32>(
          [&](int kt) { return load_a_frag(zrow_l, kt * 32, s); },
          [&](int kt) { return *(const v16h*)(wz_base + (size_t)kt * 64 * 512); });

      // xcur in C-fragment layout: one 16B load per lane
      const size_t mtg = (size_t)t * 8 + mtile;
      v8h xc = *(const v8h*)(xcur + ((mtg * 64 + ntile) * 32 + lane) * 8);
#pragma unroll
      for (int r = 0; r < 8; ++r) {
        float cur = acc[r] + (float)xc[r];
        float un = u[r] + DT * (bb * u[r] - om * v[r] + cur);
        float vn = v[r] + DT * (om * u[r] + bb * v[r]);
        u[r] = un; v[r] = vn;
      }
      // publish z_{t+1} into the OTHER buffer (no reader conflict)
#pragma unroll
      for (int r = 0; r < 8; ++r) {
        int m = mrow + r + 8 * s;
        bool sp = u[r] > THETA;
        cnt += sp ? 1u : 0u;
        zdst[(size_t)m * H + n] = sp ? (_Float16)1.0f : (_Float16)0.0f;
      }
    } else if (t > 0) {
      // readout of step t-1 uses z_t
      const _Float16* zrow_g = zsrc + (size_t)(mrow + col) * H;
      v8f acc = gemm_pipe<H / 32>(
          [&](int kt) { return load_a_frag(zrow_g, kt * 32, s); },
          [&](int kt) { return *(const v16h*)(wo_base + (size_t)kt * 2 * 512); });
#pragma unroll
      for (int r = 0; r < 8; ++r) {
        ou[r] = alpha * ou[r] + oma * acc[r];
        int m = mrow + r + 8 * s;
        out[OUTS_OFF + (size_t)(t - 1) * B * O + (size_t)m * O + n] = ou[r];
      }
    }
    // one sync per step: z_{t+1} visible; all reads of z_t drained
    grid_sync(arrive, release, (unsigned)nblocks, &phase);
  }

  // ---- tail: final readout with z_T, final state, spike total ----
  if (!is_cur) {
    const _Float16* zrow_g = zb + (size_t)(T & 1) * ZELEMS + (size_t)(mrow + col) * H;
    v8f acc = gemm_pipe<H / 32>(
        [&](int kt) { return load_a_frag(zrow_g, kt * 32, s); },
        [&](int kt) { return *(const v16h*)(wo_base + (size_t)kt * 2 * 512); });
#pragma unroll
    for (int r = 0; r < 8; ++r) {
      ou[r] = alpha * ou[r] + oma * acc[r];
      int m = mrow + r + 8 * s;
      out[OUTS_OFF + (size_t)(T - 1) * B * O + (size_t)m * O + n] = ou[r];
      out[OUF_OFF + (size_t)m * O + n] = ou[r];
    }
  } else {
#pragma unroll
    for (int r = 0; r < 8; ++r) {
      int m = mrow + r + 8 * s;
      out[ZF_OFF + (size_t)m * H + n] = (u[r] > THETA) ? 1.0f : 0.0f;
      out[UF_OFF + (size_t)m * H + n] = u[r];
    }
    atomicAdd(spikecnt, cnt);  // integer -> order-independent, deterministic
  }
  grid_sync(arrive, release, (unsigned)nblocks, &phase);
  if (blockIdx.x == 0 && threadIdx.x == 0)
    out[SSUM_OFF] = (float)(*(volatile unsigned*)spikecnt);
}

// ---------------- host ----------------
extern "C" void kernel_launch(void* const* d_in, const int* in_sizes, int n_in,
                              void* d_out, int out_size, void* d_ws, size_t ws_size,
                              hipStream_t stream) {
  const float* x        = (const float*)d_in[0];  // [T,B,I]
  const float* W_h      = (const float*)d_in[1];  // [I+H, H]
  const float* omega    = (const float*)d_in[2];  // [H]
  const float* b_offset = (const float*)d_in[3];  // [H]
  const float* W_o      = (const float*)d_in[4];  // [H, O]
  const float* tau      = (const float*)d_in[5];  // [O]
  float* out = (float*)d_out;

  char* ws = (char*)d_ws;
  unsigned*  sync_words = (unsigned*)(ws + SYNC_OFF);
  _Float16*  zbuf = (_Float16*)(ws + ZBUF_OFF);
  _Float16*  Wxp  = (_Float16*)(ws + WXP_OFF);
  _Float16*  Wzp  = (_Float16*)(ws + WZP_OFF);
  _Float16*  Wop  = (_Float16*)(ws + WOP_OFF);
  _Float16*  xcur = (_Float16*)(ws + XCUR_OFF);

  // replay-safe re-init: barrier words + both z ping-pong buffers = 0
  hipMemsetAsync(ws + SYNC_OFF, 0, 256, stream);
  hipMemsetAsync(ws + ZBUF_OFF, 0, ZBUF_BYTES, stream);

  // pack weights into WMMA B-operand layout (f16)
  {
    int totalWx = (I / 32) * 64 * 512;      // 262144
    int totalWz = (H / 32) * 64 * 512;      // 1048576
    int totalWo = (H / 32) * 2 * 512;       // 32768
    pack_w_kernel<<<(totalWx + 255) / 256, 256, 0, stream>>>(W_h, Wxp, 0, 64, totalWx, H);
    pack_w_kernel<<<(totalWz + 255) / 256, 256, 0, stream>>>(W_h, Wzp, I, 64, totalWz, H);
    pack_w_kernel<<<(totalWo + 255) / 256, 256, 0, stream>>>(W_o, Wop, 0, 2, totalWo, O);
  }

  // hoisted feed-forward GEMM: xcur[t,b,:] = x_t @ W_x  (C-fragment layout)
  xcur_gemm_kernel<<<(T * B) / 16, 256, 0, stream>>>(x, Wxp, xcur);

  // persistent recurrent core: 64 cur blocks + 2 readout blocks
  rf_recurrent_kernel<<<66, 256, 0, stream>>>(omega, b_offset, tau, Wzp, Wop,
                                              xcur, zbuf, out, sync_words, 66);
}
